// TransformerBlockV4_38508676776105
// MI455X (gfx1250) — matrix operand order repaired and checked by reading, not compiled, not run on hardware
//
#include <hip/hip_runtime.h>
#include <math.h>

// ---------------------------------------------------------------------------
// Problem constants (from reference): B=4, N=1024, E=2048, H=16, KVH=4,
// HD=128, HID=4096.  All GEMM dims are multiples of 128/32 -> no bounds checks.
// ---------------------------------------------------------------------------
typedef __bf16 bf;
typedef __attribute__((ext_vector_type(16))) __bf16 bf16x16;
typedef __attribute__((ext_vector_type(4)))  __bf16 bf16x4;
typedef __attribute__((ext_vector_type(8)))  float  v8f;

#define DEV static __device__ __forceinline__

constexpr int B_   = 4;
constexpr int N_   = 1024;
constexpr int E_   = 2048;
constexpr int H_   = 16;
constexpr int KVH_ = 4;
constexpr int HD_  = 128;
constexpr int HID_ = 4096;
constexpr int M_   = B_ * N_;          // 4096 token rows
constexpr int KVE_ = KVH_ * HD_;       // 512

DEV v8f v8f_zero() { v8f z = {0.f,0.f,0.f,0.f,0.f,0.f,0.f,0.f}; return z; }

DEV v8f wmma_bf(bf16x16 a, bf16x16 b, v8f c) {
  // V_WMMA_F32_16X16X32_BF16: D = A(16x32) * B(32x16) + C(16x16 f32)
  return __builtin_amdgcn_wmma_f32_16x16x32_bf16(false, a, false, b,
                                                 (short)0, c, false, false);
}

DEV float fast_rcp(float x) { return __builtin_amdgcn_rcpf(x); }
DEV float fast_rsq(float x) { return __builtin_amdgcn_rsqf(x); }

// CDNA5 async global->LDS copy (VGLOBAL GLOBAL_LOAD_ASYNC_TO_LDS_B128,
// tracked by ASYNCcnt).  lds_off = LDS byte address (low 32 bits of the
// generic shared pointer); gaddr = 64-bit global address.
DEV void async_g2l_b128(unsigned lds_off, const void* gaddr) {
  asm volatile("global_load_async_to_lds_b128 %0, %1, off"
               :: "v"(lds_off), "v"((unsigned long long)(uintptr_t)gaddr)
               : "memory");
}
DEV void wait_async0() {
  asm volatile("s_wait_asynccnt 0x0" ::: "memory");
}
DEV unsigned lds_addr(const void* p) { return (unsigned)(uintptr_t)p; }

// A-operand fragment (16xK tile, row-major, rows = M).
// ISA layout: lane<16 -> row=lane, K in {0..7,16..23}; lane>=16 -> K {8..15,24..31}
DEV bf16x16 load_a_frag(const bf* base, int ld) {
  const int lane = threadIdx.x & 31;
  const bf* p = base + (lane & 15) * ld + ((lane >> 4) << 3);
  bf16x16 f;
  ((uint4*)&f)[0] = *(const uint4*)(p);
  ((uint4*)&f)[1] = *(const uint4*)(p + 16);
  return f;
}

// B-operand fragment loaded from row-major [Nrows][K] (i.e. B^T storage).
// ISA layout: lane<16 -> col=lane, K in {0..7,16..23}; lane>=16 -> K {8..15,24..31}
DEV bf16x16 load_b_frag(const bf* base, int ld) {
  const int lane = threadIdx.x & 31;
  const bf* p = base + (lane & 15) * ld + ((lane >> 4) << 3);
  bf16x16 f;
  ((uint4*)&f)[0] = *(const uint4*)(p);
  ((uint4*)&f)[1] = *(const uint4*)(p + 16);
  return f;
}

// 16-lane reductions: C-matrix rows live in lane halves; xor masks 1,2,4,8
// never cross bit 4, so halves stay independent (wave32).
DEV float redsum16(float x) {
  x += __shfl_xor(x, 1, 32);
  x += __shfl_xor(x, 2, 32);
  x += __shfl_xor(x, 4, 32);
  x += __shfl_xor(x, 8, 32);
  return x;
}
DEV float redmax16(float x) {
  x = fmaxf(x, __shfl_xor(x, 1, 32));
  x = fmaxf(x, __shfl_xor(x, 2, 32));
  x = fmaxf(x, __shfl_xor(x, 4, 32));
  x = fmaxf(x, __shfl_xor(x, 8, 32));
  return x;
}

// ---------------------------------------------------------------------------
// Generic WMMA GEMM:  C[M,Nn] = A[M,K](bf16) * W[Nn,K]^T(bf16)
// 256 threads = 8 waves (4 along M x 2 along N); block tile 128x128, BK=32.
// Double-buffered LDS, filled with GLOBAL_LOAD_ASYNC_TO_LDS_B128: the async
// engine copies tile i+1 into the free buffer while the WMMAs run on tile i
// -- no VGPR staging (no spills), one s_wait_asynccnt + barrier per K-step.
// LDS pitch = 40 halves (80B): 20-bank row stride -> the 16 fragment rows map
// to disjoint bank groups (conflict-free ds_load_b128).
// EPI: 0 = f32 store, 1 = bf16 store, 2 = f32 resid + scale[col]*acc
// ---------------------------------------------------------------------------
constexpr int BM = 128, BN = 128, BK = 32, LDT = BK + 8;

template <int EPI>
__global__ void __launch_bounds__(256, 2)
gemm_wmma(const bf* __restrict__ A, const bf* __restrict__ Bw,
          int Nn, int K, float* __restrict__ outF, bf* __restrict__ outB,
          const float* __restrict__ resid, const float* __restrict__ scale) {
  __shared__ bf As[2][BM * LDT];
  __shared__ bf Bs[2][BN * LDT];
  const int tid  = threadIdx.x;
  const int wave = tid >> 5;
  const int wm   = wave & 3;     // 4 waves along M -> 32 rows each
  const int wn   = wave >> 2;    // 2 waves along N -> 64 cols each
  const int bm   = blockIdx.y, bn = blockIdx.x;

  v8f acc[2][4];
#pragma unroll
  for (int tm = 0; tm < 2; ++tm)
#pragma unroll
    for (int tn = 0; tn < 4; ++tn) acc[tm][tn] = v8f_zero();

  const int r0 = tid >> 2;
  const int c0 = (tid & 3) << 3;
  const size_t abase = (size_t)bm * BM * K;
  const size_t bbase = (size_t)bn * BN * K;

  auto async_tile = [&](int k0, int buf) {
#pragma unroll
    for (int p = 0; p < 2; ++p) {
      const int r = r0 + (p << 6);
      async_g2l_b128(lds_addr(&As[buf][r * LDT + c0]),
                     A + abase + (size_t)r * K + k0 + c0);
      async_g2l_b128(lds_addr(&Bs[buf][r * LDT + c0]),
                     Bw + bbase + (size_t)r * K + k0 + c0);
    }
  };

  const int NIT = K / BK;
  async_tile(0, 0);
  wait_async0();
  __syncthreads();

  for (int i = 0; i < NIT; ++i) {
    const int cur = i & 1;
    // fill the free buffer while the matrix pipe works on the current one
    if (i + 1 < NIT) async_tile((i + 1) * BK, cur ^ 1);

    bf16x16 af[2], bw[4];
#pragma unroll
    for (int tm = 0; tm < 2; ++tm)
      af[tm] = load_a_frag(&As[cur][(wm * 32 + tm * 16) * LDT], LDT);
#pragma unroll
    for (int tn = 0; tn < 4; ++tn)
      bw[tn] = load_b_frag(&Bs[cur][(wn * 64 + tn * 16) * LDT], LDT);
#pragma unroll
    for (int tm = 0; tm < 2; ++tm)
#pragma unroll
      for (int tn = 0; tn < 4; ++tn)
        acc[tm][tn] = wmma_bf(af[tm], bw[tn], acc[tm][tn]);

    if (i + 1 < NIT) wait_async0();
    __syncthreads();
  }

  const int lane = tid & 31;
  const int col  = lane & 15;
  const int rb2  = (lane >> 4) << 3;   // C layout: vgpr v -> row rb2+v
#pragma unroll
  for (int tm = 0; tm < 2; ++tm)
#pragma unroll
    for (int tn = 0; tn < 4; ++tn) {
      const int gr = bm * BM + wm * 32 + tm * 16 + rb2;
      const int gc = bn * BN + wn * 64 + tn * 16 + col;
#pragma unroll
      for (int v = 0; v < 8; ++v) {
        const size_t idx = (size_t)(gr + v) * Nn + gc;
        const float val = acc[tm][tn][v];
        if (EPI == 0)      outF[idx] = val;
        else if (EPI == 1) outB[idx] = (bf)val;
        else               outF[idx] = resid[idx] + scale[gc] * val;
      }
    }
}

// ---------------------------------------------------------------------------
// Dual-B GEMM for SwiGLU: shares the A tile between w1 and w3, epilogue
// writes g = silu(A*w1^T) * (A*w3^T) as bf16.  Block tile 128x64,
// async-double-buffered like gemm_wmma.
// ---------------------------------------------------------------------------
constexpr int BND = 64;

__global__ void __launch_bounds__(256, 2)
gemm_swiglu(const bf* __restrict__ A, const bf* __restrict__ B1,
            const bf* __restrict__ B3, bf* __restrict__ G, int Nn, int K) {
  __shared__ bf As[2][BM * LDT];
  __shared__ bf Bs1[2][BND * LDT];
  __shared__ bf Bs3[2][BND * LDT];
  const int tid  = threadIdx.x;
  const int wave = tid >> 5;
  const int wm   = wave & 3;     // 32 rows
  const int wn   = wave >> 2;    // 32 cols
  const int bm   = blockIdx.y, bn = blockIdx.x;

  v8f a1[2][2], a3[2][2];
#pragma unroll
  for (int tm = 0; tm < 2; ++tm)
#pragma unroll
    for (int tn = 0; tn < 2; ++tn) { a1[tm][tn] = v8f_zero(); a3[tm][tn] = v8f_zero(); }

  const int r0 = tid >> 2;
  const int c0 = (tid & 3) << 3;
  const size_t abase = (size_t)bm * BM * K;
  const size_t bbase = (size_t)bn * BND * K;

  auto async_tile = [&](int k0, int buf) {
#pragma unroll
    for (int p = 0; p < 2; ++p) {
      const int r = r0 + (p << 6);
      async_g2l_b128(lds_addr(&As[buf][r * LDT + c0]),
                     A + abase + (size_t)r * K + k0 + c0);
    }
    if (r0 < 64) {
      async_g2l_b128(lds_addr(&Bs1[buf][r0 * LDT + c0]),
                     B1 + bbase + (size_t)r0 * K + k0 + c0);
      async_g2l_b128(lds_addr(&Bs3[buf][r0 * LDT + c0]),
                     B3 + bbase + (size_t)r0 * K + k0 + c0);
    }
  };

  const int NIT = K / BK;
  async_tile(0, 0);
  wait_async0();
  __syncthreads();

  for (int i = 0; i < NIT; ++i) {
    const int cur = i & 1;
    if (i + 1 < NIT) async_tile((i + 1) * BK, cur ^ 1);

    bf16x16 af[2];
#pragma unroll
    for (int tm = 0; tm < 2; ++tm)
      af[tm] = load_a_frag(&As[cur][(wm * 32 + tm * 16) * LDT], LDT);
#pragma unroll
    for (int tn = 0; tn < 2; ++tn) {
      bf16x16 b1 = load_b_frag(&Bs1[cur][(wn * 32 + tn * 16) * LDT], LDT);
      bf16x16 b3 = load_b_frag(&Bs3[cur][(wn * 32 + tn * 16) * LDT], LDT);
#pragma unroll
      for (int tm = 0; tm < 2; ++tm) {
        a1[tm][tn] = wmma_bf(af[tm], b1, a1[tm][tn]);
        a3[tm][tn] = wmma_bf(af[tm], b3, a3[tm][tn]);
      }
    }

    if (i + 1 < NIT) wait_async0();
    __syncthreads();
  }

  const int lane = tid & 31;
  const int col  = lane & 15;
  const int rb2  = (lane >> 4) << 3;
#pragma unroll
  for (int tm = 0; tm < 2; ++tm)
#pragma unroll
    for (int tn = 0; tn < 2; ++tn) {
      const int gr = bm * BM + wm * 32 + tm * 16 + rb2;
      const int gc = bn * BND + wn * 32 + tn * 16 + col;
#pragma unroll
      for (int v = 0; v < 8; ++v) {
        const float c1 = a1[tm][tn][v];
        const float c3 = a3[tm][tn][v];
        // silu via v_rcp_f32 (avoid full IEEE division expansion)
        const float s = c1 * fast_rcp(1.f + __expf(-c1));
        G[(size_t)(gr + v) * Nn + gc] = (bf)(s * c3);
      }
    }
}

// ---------------------------------------------------------------------------
// Flash attention, one block = 128 q rows of one (b, h).  8 waves x 16 rows.
// K-tile (64x128) and transposed V-tile (128x64) staged in LDS (shared by all
// waves); P re-staged per-wave to re-fragment as WMMA A operand.
// Epilogue fuses 1/l and the "xsa" per-position v-projection removal.
// ---------------------------------------------------------------------------
constexpr int KT  = 64;
constexpr int LDK = HD_ + 8;   // 136 halves / 272B rows -> conflict-free frags
constexpr int LDV = KT + 8;    // 72 halves / 144B rows

__global__ void __launch_bounds__(256, 2)
attn_fwd(const bf* __restrict__ q, const bf* __restrict__ k,
         const bf* __restrict__ v, bf* __restrict__ y) {
  __shared__ bf Ks[KT * LDK];
  __shared__ bf Vt[HD_ * LDV];
  __shared__ bf Ps[8 * 16 * LDV];

  const int tid  = threadIdx.x;
  const int wave = tid >> 5;
  const int lane = tid & 31;
  const int col  = lane & 15;
  const int rb   = (lane >> 4) << 3;

  const int bh  = blockIdx.y;
  const int b   = bh / H_;
  const int h   = bh - b * H_;
  const int kvh = h >> 2;                       // H/KVH = 4 (jnp.repeat)
  const int qrow0 = blockIdx.x * 128 + wave * 16;

  const bf* qbase = q + ((size_t)b * N_) * E_   + (size_t)h   * HD_;
  const bf* kbase = k + ((size_t)b * N_) * KVE_ + (size_t)kvh * HD_;
  const bf* vbase = v + ((size_t)b * N_) * KVE_ + (size_t)kvh * HD_;

  bf16x16 qf[4];                                // Q rows live in registers
#pragma unroll
  for (int ks = 0; ks < 4; ++ks)
    qf[ks] = load_a_frag(qbase + (size_t)qrow0 * E_ + ks * 32, E_);

  v8f acc[8];
#pragma unroll
  for (int dt = 0; dt < 8; ++dt) acc[dt] = v8f_zero();
  float l[8], mx[8];
#pragma unroll
  for (int i = 0; i < 8; ++i) { l[i] = 0.f; mx[i] = -__builtin_inff(); }

  const float sscale = 0.088388347648318447f;   // 128^-0.5

  for (int kt = 0; kt < N_; kt += KT) {
    __syncthreads();
    // cooperative load: K tile row-major, V tile transposed into [d][k]
#pragma unroll
    for (int i = 0; i < 2; ++i) {
      const int cid = tid + (i << 8);
      const int r = cid >> 3;
      const int c = (cid & 7) << 3;
      *(uint4*)(Ks + r * LDK + c) =
          *(const uint4*)(kbase + (size_t)(kt + r) * KVE_ + c);
      union { uint4 u; bf e[8]; } dv;
      dv.u = *(const uint4*)(vbase + (size_t)(kt + r) * KVE_ + c);
#pragma unroll
      for (int j = 0; j < 8; ++j) Vt[(c + j) * LDV + r] = dv.e[j];
    }
    __syncthreads();

    // S = Q K^T  (B operand = K rows, contraction over d)
    v8f s[4];
#pragma unroll
    for (int tn = 0; tn < 4; ++tn) {
      s[tn] = v8f_zero();
#pragma unroll
      for (int ks = 0; ks < 4; ++ks) {
        bf16x16 bk = load_b_frag(Ks + (tn * 16) * LDK + ks * 32, LDK);
        s[tn] = wmma_bf(qf[ks], bk, s[tn]);
      }
    }
#pragma unroll
    for (int tn = 0; tn < 4; ++tn)
#pragma unroll
      for (int vv = 0; vv < 8; ++vv) s[tn][vv] *= sscale;

    // online softmax (row = rb+vv of this wave's 16 q rows)
    float alpha[8];
#pragma unroll
    for (int vv = 0; vv < 8; ++vv) {
      float m0 = fmaxf(fmaxf(s[0][vv], s[1][vv]), fmaxf(s[2][vv], s[3][vv]));
      m0 = redmax16(m0);
      const float nm = fmaxf(mx[vv], m0);
      alpha[vv] = __expf(mx[vv] - nm);
      mx[vv] = nm;
    }
#pragma unroll
    for (int tn = 0; tn < 4; ++tn)
#pragma unroll
      for (int vv = 0; vv < 8; ++vv) s[tn][vv] = __expf(s[tn][vv] - mx[vv]);
#pragma unroll
    for (int vv = 0; vv < 8; ++vv) {
      float rs = s[0][vv] + s[1][vv] + s[2][vv] + s[3][vv];
      rs = redsum16(rs);
      l[vv] = l[vv] * alpha[vv] + rs;
    }
#pragma unroll
    for (int dt = 0; dt < 8; ++dt)
#pragma unroll
      for (int vv = 0; vv < 8; ++vv) acc[dt][vv] *= alpha[vv];

    // stage P (C layout -> row-major LDS) and re-fragment as A operand
    bf* myP = Ps + wave * 16 * LDV;
#pragma unroll
    for (int tn = 0; tn < 4; ++tn)
#pragma unroll
      for (int vv = 0; vv < 8; ++vv)
        myP[(rb + vv) * LDV + tn * 16 + col] = (bf)s[tn][vv];
    __syncthreads();

    bf16x16 pa[2];
#pragma unroll
    for (int ks2 = 0; ks2 < 2; ++ks2)
      pa[ks2] = load_a_frag(myP + ks2 * 32, LDV);
#pragma unroll
    for (int dt = 0; dt < 8; ++dt)
#pragma unroll
      for (int ks2 = 0; ks2 < 2; ++ks2) {
        bf16x16 bv = load_b_frag(Vt + (dt * 16) * LDV + ks2 * 32, LDV);
        acc[dt] = wmma_bf(pa[ks2], bv, acc[dt]);
      }
  }

  // epilogue: 1/l, xsa removal (project out v at the query's own position)
#pragma unroll
  for (int vv = 0; vv < 8; ++vv) {
    const float inv = fast_rcp(l[vv]);
    const int qp = qrow0 + rb + vv;
    const bf* vrow = vbase + (size_t)qp * KVE_;
    float vr[8], dot = 0.f, vsq = 0.f;
#pragma unroll
    for (int dt = 0; dt < 8; ++dt) {
      acc[dt][vv] *= inv;
      vr[dt] = (float)vrow[dt * 16 + col];
      dot += acc[dt][vv] * vr[dt];
      vsq += vr[dt] * vr[dt];
    }
    dot = redsum16(dot);
    vsq = redsum16(vsq);
    // == (y.vn)/max(|v|,1e-12) with vn=v/max(|v|,1e-12)
    const float coef = dot * fast_rcp(fmaxf(vsq, 1e-24f));
    bf* yrow = y + ((size_t)b * N_ + qp) * E_ + (size_t)h * HD_;
#pragma unroll
    for (int dt = 0; dt < 8; ++dt)
      yrow[dt * 16 + col] = (bf)(acc[dt][vv] - coef * vr[dt]);
  }
}

// ---------------------------------------------------------------------------
// Elementwise / norm kernels
// ---------------------------------------------------------------------------
__global__ void cvt_bf16_k(const float* __restrict__ in, bf* __restrict__ out,
                           int n4) {
  const int i = blockIdx.x * 256 + threadIdx.x;
  if (i < n4) {
    const float4 f = ((const float4*)in)[i];
    bf16x4 o; o.x = (bf)f.x; o.y = (bf)f.y; o.z = (bf)f.z; o.w = (bf)f.w;
    ((bf16x4*)out)[i] = o;
  }
}

// x_routed = rm0*x + rm1*x0; out = rms_norm(x_routed, norm1_w) as bf16
__global__ void __launch_bounds__(256)
route_norm1(const float* __restrict__ x, const float* __restrict__ x0,
            const float* __restrict__ rmix, const float* __restrict__ w,
            bf* __restrict__ out) {
  const int row = blockIdx.x, tid = threadIdx.x;
  const float* px = x  + (size_t)row * E_;
  const float* p0 = x0 + (size_t)row * E_;
  float vals[8], ss = 0.f;
#pragma unroll
  for (int j = 0; j < 8; ++j) {
    const int e = (j << 8) + tid;
    const float xr = rmix[e] * px[e] + rmix[E_ + e] * p0[e];
    vals[j] = xr; ss += xr * xr;
  }
#pragma unroll
  for (int m = 1; m < 32; m <<= 1) ss += __shfl_xor(ss, m, 32);
  __shared__ float red[8];
  if ((tid & 31) == 0) red[tid >> 5] = ss;
  __syncthreads();
  float tot = 0.f;
#pragma unroll
  for (int i = 0; i < 8; ++i) tot += red[i];
  const float sc = fast_rsq(tot * (1.f / E_) + 1e-6f);
  bf* po = out + (size_t)row * E_;
#pragma unroll
  for (int j = 0; j < 8; ++j) {
    const int e = (j << 8) + tid;
    po[e] = (bf)(vals[j] * sc * w[e]);
  }
}

// generic row rms_norm over E; OUTBF: bf16 out, else f32 out (in-place safe)
template <bool OUTBF>
__global__ void __launch_bounds__(256)
rmsnorm_k(const float* __restrict__ in, const float* __restrict__ w, float eps,
          bf* __restrict__ outb, float* __restrict__ outf) {
  const int row = blockIdx.x, tid = threadIdx.x;
  const float* p = in + (size_t)row * E_;
  float vals[8], ss = 0.f;
#pragma unroll
  for (int j = 0; j < 8; ++j) {
    const int e = (j << 8) + tid;
    const float x = p[e];
    vals[j] = x; ss += x * x;
  }
#pragma unroll
  for (int m = 1; m < 32; m <<= 1) ss += __shfl_xor(ss, m, 32);
  __shared__ float red[8];
  if ((tid & 31) == 0) red[tid >> 5] = ss;
  __syncthreads();
  float tot = 0.f;
#pragma unroll
  for (int i = 0; i < 8; ++i) tot += red[i];
  const float sc = fast_rsq(tot * (1.f / E_) + eps);
#pragma unroll
  for (int j = 0; j < 8; ++j) {
    const int e = (j << 8) + tid;
    const float o = vals[j] * sc * w[e];
    if (OUTBF) outb[(size_t)row * E_ + e] = (bf)o;
    else       outf[(size_t)row * E_ + e] = o;
  }
}

// per-head (128-wide) rms_norm, no weight, eps = f32 machine eps; f32 -> bf16.
// rows of [*, C] with C % 128 == 0, so blocks of 128 elements align to heads.
__global__ void __launch_bounds__(128)
qk_norm_k(const float* __restrict__ in, bf* __restrict__ out) {
  const size_t base = (size_t)blockIdx.x * 128;
  const int tid = threadIdx.x;
  const float x = in[base + tid];
  float ss = x * x;
#pragma unroll
  for (int m = 1; m < 32; m <<= 1) ss += __shfl_xor(ss, m, 32);
  __shared__ float red[4];
  if ((tid & 31) == 0) red[tid >> 5] = ss;
  __syncthreads();
  const float tot = red[0] + red[1] + red[2] + red[3];
  const float sc = fast_rsq(tot * (1.f / 128.f) + 1.1920929e-07f);
  out[base + tid] = (bf)(x * sc);
}

// ---------------------------------------------------------------------------
// Workspace layout (bytes).  Peak ~196 MiB with phase aliasing: the MLP
// buffers overlay q/k buffers that are dead after attention + O-projection.
// ---------------------------------------------------------------------------
constexpr size_t SZ_WQ = (size_t)E_ * E_ * 2;
constexpr size_t SZ_WK = (size_t)KVE_ * E_ * 2;
constexpr size_t SZ_W1 = (size_t)HID_ * E_ * 2;
constexpr size_t OFF_WQ = 0;
constexpr size_t OFF_WK = OFF_WQ + SZ_WQ;
constexpr size_t OFF_WV = OFF_WK + SZ_WK;
constexpr size_t OFF_WO = OFF_WV + SZ_WK;
constexpr size_t OFF_W1 = OFF_WO + SZ_WQ;
constexpr size_t OFF_W2 = OFF_W1 + SZ_W1;
constexpr size_t OFF_W3 = OFF_W2 + SZ_W1;
constexpr size_t OFF_HB = OFF_W3 + SZ_W1;                    // bf16 [M,E]
constexpr size_t OFF_QF = OFF_HB + (size_t)M_ * E_ * 2;      // f32  [M,E]
constexpr size_t OFF_KF = OFF_QF + (size_t)M_ * E_ * 4;      // f32  [M,512]
constexpr size_t OFF_QB = OFF_KF + (size_t)M_ * KVE_ * 4;    // bf16 [M,E]
constexpr size_t OFF_KB = OFF_QB + (size_t)M_ * E_ * 2;      // bf16 [M,512]
constexpr size_t OFF_VB = OFF_KB + (size_t)M_ * KVE_ * 2;    // bf16 [M,512]
constexpr size_t OFF_YB = OFF_VB + (size_t)M_ * KVE_ * 2;    // bf16 [M,E]
constexpr size_t OFF_X1 = OFF_YB + (size_t)M_ * E_ * 2;      // f32  [M,E]
constexpr size_t WS_NEED = OFF_X1 + (size_t)M_ * E_ * 4;
// phase-2 aliases (q/k buffers dead after attention):
constexpr size_t OFF_MB = OFF_QF;                            // bf16 [M,E]
constexpr size_t OFF_G  = OFF_QF + (size_t)M_ * E_ * 2;      // bf16 [M,HID]

extern "C" void kernel_launch(void* const* d_in, const int* in_sizes, int n_in,
                              void* d_out, int out_size, void* d_ws,
                              size_t ws_size, hipStream_t stream) {
  (void)in_sizes; (void)n_in; (void)out_size;
  if (ws_size < WS_NEED) return;

  const float* x      = (const float*)d_in[0];
  const float* x0     = (const float*)d_in[1];
  const float* n1w    = (const float*)d_in[2];
  const float* Wq     = (const float*)d_in[3];
  const float* Wk     = (const float*)d_in[4];
  const float* Wv     = (const float*)d_in[5];
  const float* Wo     = (const float*)d_in[6];
  const float* n2w    = (const float*)d_in[7];
  const float* w1     = (const float*)d_in[8];
  const float* w2     = (const float*)d_in[9];
  const float* w3     = (const float*)d_in[10];
  const float* ascale = (const float*)d_in[11];
  const float* mscale = (const float*)d_in[12];
  const float* rmix   = (const float*)d_in[13];
  const float* pnw    = (const float*)d_in[14];
  float* out = (float*)d_out;
  char*  ws  = (char*)d_ws;

  bf* wqb = (bf*)(ws + OFF_WQ); bf* wkb = (bf*)(ws + OFF_WK);
  bf* wvb = (bf*)(ws + OFF_WV); bf* wob = (bf*)(ws + OFF_WO);
  bf* w1b = (bf*)(ws + OFF_W1); bf* w2b = (bf*)(ws + OFF_W2);
  bf* w3b = (bf*)(ws + OFF_W3);
  bf* hb  = (bf*)(ws + OFF_HB);
  float* qfp = (float*)(ws + OFF_QF);
  float* kfp = (float*)(ws + OFF_KF);
  bf* qb  = (bf*)(ws + OFF_QB); bf* kb = (bf*)(ws + OFF_KB);
  bf* vb  = (bf*)(ws + OFF_VB); bf* yb = (bf*)(ws + OFF_YB);
  float* x1 = (float*)(ws + OFF_X1);
  bf* mb  = (bf*)(ws + OFF_MB); bf* g  = (bf*)(ws + OFF_G);

  auto cvt = [&](const float* s, bf* d, size_t n) {
    cvt_bf16_k<<<(unsigned)(n / 4 / 256), 256, 0, stream>>>(s, d, (int)(n / 4));
  };
  // 1) weights -> bf16
  cvt(Wq, wqb, (size_t)E_ * E_);
  cvt(Wk, wkb, (size_t)KVE_ * E_);
  cvt(Wv, wvb, (size_t)KVE_ * E_);
  cvt(Wo, wob, (size_t)E_ * E_);
  cvt(w1, w1b, (size_t)HID_ * E_);
  cvt(w2, w2b, (size_t)HID_ * E_);
  cvt(w3, w3b, (size_t)HID_ * E_);

  // 2) residual routing + rmsnorm1 -> bf16 activations
  route_norm1<<<M_, 256, 0, stream>>>(x, x0, rmix, n1w, hb);

  // 3) QKV projections (q/k as f32 for head-norm, v directly bf16)
  gemm_wmma<0><<<dim3(E_ / BM, M_ / BM), 256, 0, stream>>>(
      hb, wqb, E_, E_, qfp, nullptr, nullptr, nullptr);
  gemm_wmma<0><<<dim3(KVE_ / BM, M_ / BM), 256, 0, stream>>>(
      hb, wkb, KVE_, E_, kfp, nullptr, nullptr, nullptr);
  gemm_wmma<1><<<dim3(KVE_ / BM, M_ / BM), 256, 0, stream>>>(
      hb, wvb, KVE_, E_, nullptr, vb, nullptr, nullptr);

  // 4) per-head rms norm of q, k (eps = f32 machine eps)
  qk_norm_k<<<(unsigned)((size_t)M_ * E_ / 128), 128, 0, stream>>>(qfp, qb);
  qk_norm_k<<<(unsigned)((size_t)M_ * KVE_ / 128), 128, 0, stream>>>(kfp, kb);

  // 5) flash attention with fused xsa removal -> y (bf16, [B,N,E] layout)
  attn_fwd<<<dim3(N_ / 128, B_ * H_), 256, 0, stream>>>(qb, kb, vb, yb);

  // 6) O projection, fused x + attn_scale * C
  gemm_wmma<2><<<dim3(E_ / BM, M_ / BM), 256, 0, stream>>>(
      yb, wob, E_, E_, x1, nullptr, x, ascale);

  // 7) rmsnorm2 -> bf16
  rmsnorm_k<true><<<M_, 256, 0, stream>>>(x1, n2w, 1e-6f, mb, nullptr);

  // 8) dual GEMM + SwiGLU -> g (bf16)
  gemm_swiglu<<<dim3(HID_ / BND, M_ / BM), 256, 0, stream>>>(
      mb, w1b, w3b, g, HID_, E_);

  // 9) w2 projection, fused x1 + mlp_scale * C -> d_out (f32)
  gemm_wmma<2><<<dim3(E_ / BM, M_ / BM), 256, 0, stream>>>(
      g, w2b, E_, HID_, out, nullptr, x1, mscale);

  // 10) post norm, in-place on d_out
  rmsnorm_k<false><<<M_, 256, 0, stream>>>(out, pnw, 1e-6f, nullptr, out);
}
